// Attention_33595234189924
// MI455X (gfx1250) — matrix operand order repaired and checked by reading, not compiled, not run on hardware
//
#include <hip/hip_runtime.h>

typedef __attribute__((ext_vector_type(16))) _Float16 v16h;
typedef __attribute__((ext_vector_type(8)))  _Float16 v8h;
typedef __attribute__((ext_vector_type(8)))  float    v8f;

constexpr int B_  = 2;
constexpr int S_  = 2048;
constexpr int D_  = 2048;
constexpr int H_  = 16;
constexpr int HD_ = 128;

// ---------------------------------------------------------------------------
// Fragment loaders per CDNA5 ISA VGPR layouts (CDNA5 ISA §7.12.2)
// A (16x32 f16): lane L -> row M=L%16; VGPR0-3 hold K = khi..khi+7,
//                VGPR4-7 hold K = 16+khi..16+khi+7, khi = (L>=16)?8:0.
// B (32x16 f16): lane L -> col N=L%16; VGPR0-3 hold K = kb..kb+7,
//                VGPR4-7 hold K = 16+kb..16+kb+7, kb = (L>=16)?8:0.  Element B[k][n] = src[n*ld + k].
// C/D (16x16 f32): lane L -> col N=L%16; VGPR r -> row M = r + ((L>=16)?8:0).
// ---------------------------------------------------------------------------
__device__ __forceinline__ v16h load_a_frag(const _Float16* base, int ld) {
  const int lane = threadIdx.x & 31;
  const int row  = lane & 15;
  const int khi  = (lane >> 4) << 3;                 // 0 or 8
  const _Float16* p = base + row * ld + khi;
  v8h lo = *(const v8h*)(p);                         // K = khi .. khi+7
  v8h hi = *(const v8h*)(p + 16);                    // K = 16+khi .. 16+khi+7
  v16h a;
#pragma unroll
  for (int i = 0; i < 8; ++i) { a[i] = lo[i]; a[i + 8] = hi[i]; }
  return a;
}

__device__ __forceinline__ v16h load_b_frag(const _Float16* src, int ld) {
  const int lane  = threadIdx.x & 31;
  const int col   = lane & 15;
  const int kbase = (lane >> 4) << 3;                // 0 or 8
  const _Float16* p = src + col * ld + kbase;
  v8h b0 = *(const v8h*)(p);                         // K = kbase .. kbase+7
  v8h b1 = *(const v8h*)(p + 16);                    // K = 16+kbase .. 16+kbase+7
  v16h b;
#pragma unroll
  for (int i = 0; i < 8; ++i) { b[i] = b0[i]; b[i + 8] = b1[i]; }
  return b;
}

// ---------------------------------------------------------------------------
// fp32 -> f16 conversion (inputs are fp32; WMMA path wants f16)
// ---------------------------------------------------------------------------
__global__ void cvt_f32_f16(const float* __restrict__ in,
                            _Float16* __restrict__ out, int n) {
  int i = blockIdx.x * blockDim.x + threadIdx.x;
  if (i < n) out[i] = (_Float16)in[i];
}

// ---------------------------------------------------------------------------
// Fused projection GEMM:  out = X @ W^T + b   (+ RoPE for Q/K)
// Register-blocked: each wave owns a 64x64 macro-tile = 4x4 grid of 16x16
// WMMA tiles. One k-step (K=32): 4 A-frags + 4 B-frags (16 x b128) feed
// 16 v_wmma -> 1 wmma per b128 issue, 4x the reuse of the naive mapping.
// Block = 8 waves tiled 2(M) x 4(N) -> 128x256 per workgroup.
//  Q/K output: [B*H, S, HD] f16 (RoPE applied on f32 accumulators)
//  V  output:  [B*H, HD, S] f16 (transposed so PV B-frags are b128 loads)
// ---------------------------------------------------------------------------
__global__ __launch_bounds__(256) void qkv_gemm_kernel(
    const _Float16* __restrict__ X, const _Float16* __restrict__ W,
    const float* __restrict__ bias, const float* __restrict__ freqs,
    _Float16* __restrict__ out, int apply_rope, int transpose_out) {
  const int wave   = threadIdx.x >> 5;
  const int lane   = threadIdx.x & 31;
  const int wave_m = wave >> 2;                       // 0..1
  const int wave_n = wave & 3;                        // 0..3
  const int m_wave = blockIdx.y * 128 + wave_m * 64;
  const int n_wave = blockIdx.x * 256 + wave_n * 64;

  v8f acc[4][4] = {};
  for (int kk = 0; kk < D_; kk += 32) {
    v16h a[4], b[4];
#pragma unroll
    for (int i = 0; i < 4; ++i)
      a[i] = load_a_frag(X + (size_t)(m_wave + i * 16) * D_ + kk, D_);
#pragma unroll
    for (int j = 0; j < 4; ++j)
      b[j] = load_b_frag(W + (size_t)(n_wave + j * 16) * D_ + kk, D_);
#pragma unroll
    for (int i = 0; i < 4; ++i)
#pragma unroll
      for (int j = 0; j < 4; ++j)
        acc[i][j] = __builtin_amdgcn_wmma_f32_16x16x32_f16(
            false, a[i], false, b[j], (short)0, acc[i][j], false, false);
  }

  const int col = lane & 15;
  const int hi8 = (lane >> 4) << 3;

#pragma unroll
  for (int j = 0; j < 4; ++j) {
    const int n  = n_wave + j * 16 + col;
    const int h  = n >> 7;                 // n / HD
    const int hd = n & (HD_ - 1);
    const float bn = bias[n];
#pragma unroll
    for (int i = 0; i < 4; ++i) {
      const int m0 = m_wave + i * 16 + hi8;
      float vals[8];
#pragma unroll
      for (int r = 0; r < 8; ++r) vals[r] = acc[i][j][r] + bn;

      if (apply_rope) {
        // Complex rotate on (even,odd) hd pairs; partner sits in lane^1.
#pragma unroll
        for (int r = 0; r < 8; ++r) {
          const int s = (m0 + r) & (S_ - 1);
          const float ang = freqs[s * (HD_ / 2) + (hd >> 1)];
          const float c = __cosf(ang), sn = __sinf(ang);
          const float partner = __shfl_xor(vals[r], 1);
          vals[r] = vals[r] * c + partner * ((hd & 1) ? sn : -sn);
        }
      }

      const int b_idx = m0 >> 11;          // m0 / S
      const int bh    = b_idx * H_ + h;
      if (transpose_out) {                 // V: [bh, hd, s], 8 contiguous s
        v8h pk;
#pragma unroll
        for (int r = 0; r < 8; ++r) pk[r] = (_Float16)vals[r];
        const int s0 = m0 & (S_ - 1);
        *(v8h*)(out + ((size_t)bh * HD_ + hd) * S_ + s0) = pk;
      } else {                             // Q/K: [bh, s, hd]
#pragma unroll
        for (int r = 0; r < 8; ++r) {
          const int s = (m0 + r) & (S_ - 1);
          out[((size_t)bh * S_ + s) * HD_ + hd] = (_Float16)vals[r];
        }
      }
    }
  }
}

// ---------------------------------------------------------------------------
// Flash attention, one wave per 16 query rows, 32-key steps.
// Scores: 8 WMMAs (2 n-tiles x 4 hd-chunks); online softmax with shfl
// reductions (rows live in 16-lane half-groups); P staged via per-wave LDS
// (C-layout -> A-layout crossing lanes) guarded by s_wait_dscnt; PV: 8 WMMAs
// against V^T b128 B-fragments. O (16x128 f32) lives in VGPRs.
// ---------------------------------------------------------------------------
__global__ __launch_bounds__(256) void flash_attn_kernel(
    const _Float16* __restrict__ Q,   // [B*H, S, HD]
    const _Float16* __restrict__ K,   // [B*H, S, HD]
    const _Float16* __restrict__ Vt,  // [B*H, HD, S]
    const float* __restrict__ mask,   // [B, 1, S, S]
    float* __restrict__ out) {        // [B, S, D]
  __shared__ __align__(16) _Float16 pbuf[8 * 16 * 32];
  const int wave = threadIdx.x >> 5;
  const int lane = threadIdx.x & 31;
  const int bh   = blockIdx.x >> 4;
  const int q0   = ((blockIdx.x & 15) * 8 + wave) * 16;
  const int b    = bh >> 4;
  const int h    = bh & 15;

  const _Float16* Qp = Q + (size_t)bh * S_ * HD_;
  const _Float16* Kp = K + (size_t)bh * S_ * HD_;
  const _Float16* Vp = Vt + (size_t)bh * HD_ * S_;
  const float*    mp = mask + (size_t)b * S_ * S_;
  _Float16* pl = pbuf + wave * (16 * 32);

  v16h aq[4];
#pragma unroll
  for (int c = 0; c < 4; ++c)
    aq[c] = load_a_frag(Qp + (size_t)q0 * HD_ + c * 32, HD_);

  v8f o[8] = {};
  float mrow[8], lrow[8];
#pragma unroll
  for (int r = 0; r < 8; ++r) { mrow[r] = -1e30f; lrow[r] = 0.f; }

  const int col = lane & 15;
  const int hi8 = (lane >> 4) << 3;
  const float sc = 0.088388347648318447f;  // 1/sqrt(HD)

  for (int kt = 0; kt < S_; kt += 32) {
    if (kt + 32 < S_)
      __builtin_prefetch(Kp + (size_t)(kt + 32) * HD_, 0, 1);

    v8f s0 = {}, s1 = {};
#pragma unroll
    for (int c = 0; c < 4; ++c) {
      v16h bk = load_b_frag(Kp + (size_t)kt * HD_ + c * 32, HD_);
      s0 = __builtin_amdgcn_wmma_f32_16x16x32_f16(false, aq[c], false, bk,
                                                  (short)0, s0, false, false);
    }
#pragma unroll
    for (int c = 0; c < 4; ++c) {
      v16h bk = load_b_frag(Kp + (size_t)(kt + 16) * HD_ + c * 32, HD_);
      s1 = __builtin_amdgcn_wmma_f32_16x16x32_f16(false, aq[c], false, bk,
                                                  (short)0, s1, false, false);
    }

#pragma unroll
    for (int r = 0; r < 8; ++r) {
      const int qrow = q0 + hi8 + r;
      float v0 = s0[r] * sc + mp[(size_t)qrow * S_ + kt + col];
      float v1 = s1[r] * sc + mp[(size_t)qrow * S_ + kt + 16 + col];
      float rmax = fmaxf(v0, v1);
      rmax = fmaxf(rmax, __shfl_xor(rmax, 1));
      rmax = fmaxf(rmax, __shfl_xor(rmax, 2));
      rmax = fmaxf(rmax, __shfl_xor(rmax, 4));
      rmax = fmaxf(rmax, __shfl_xor(rmax, 8));
      const float mnew  = fmaxf(mrow[r], rmax);
      const float alpha = __expf(mrow[r] - mnew);
      const float p0 = __expf(v0 - mnew);
      const float p1 = __expf(v1 - mnew);
      float rs = p0 + p1;
      rs += __shfl_xor(rs, 1);
      rs += __shfl_xor(rs, 2);
      rs += __shfl_xor(rs, 4);
      rs += __shfl_xor(rs, 8);
      lrow[r] = lrow[r] * alpha + rs;
      mrow[r] = mnew;
#pragma unroll
      for (int t = 0; t < 8; ++t) o[t][r] *= alpha;
      const int prow = hi8 + r;                 // C-layout -> LDS row-major
      pl[prow * 32 + col]      = (_Float16)p0;
      pl[prow * 32 + 16 + col] = (_Float16)p1;
    }

    // Cross-lane LDS exchange: make P stores visible before A-frag reads.
    asm volatile("s_wait_dscnt 0" ::: "memory");
    v16h ap = load_a_frag(pl, 32);
#pragma unroll
    for (int t = 0; t < 8; ++t) {
      v16h bv = load_b_frag(Vp + (size_t)(t * 16) * S_ + kt, S_);
      o[t] = __builtin_amdgcn_wmma_f32_16x16x32_f16(false, ap, false, bv,
                                                    (short)0, o[t], false, false);
    }
    asm volatile("s_wait_dscnt 0" ::: "memory");  // reads done before overwrite
  }

#pragma unroll
  for (int r = 0; r < 8; ++r) {
    const float inv = 1.f / lrow[r];
    const int qrow = q0 + hi8 + r;
    float* op = out + ((size_t)b * S_ + qrow) * D_ + h * HD_;
#pragma unroll
    for (int t = 0; t < 8; ++t) op[t * 16 + col] = o[t][r] * inv;
  }
}

// ---------------------------------------------------------------------------
// inputs: sequence, frequencies, mask, Wq, bq, Wk, bk, Wv, bv
// workspace (f16): Xh[8M] Wqh[4M] Wkh[4M] Wvh[4M] Qh[8M] Kh[8M] Vt[8M] = 88 MB
// ---------------------------------------------------------------------------
extern "C" void kernel_launch(void* const* d_in, const int* in_sizes, int n_in,
                              void* d_out, int out_size, void* d_ws,
                              size_t ws_size, hipStream_t stream) {
  (void)in_sizes; (void)n_in; (void)out_size; (void)ws_size;
  const float* seq   = (const float*)d_in[0];
  const float* freqs = (const float*)d_in[1];
  const float* mask  = (const float*)d_in[2];
  const float* Wq    = (const float*)d_in[3];
  const float* bq    = (const float*)d_in[4];
  const float* Wk    = (const float*)d_in[5];
  const float* bk    = (const float*)d_in[6];
  const float* Wv    = (const float*)d_in[7];
  const float* bv    = (const float*)d_in[8];
  float* out = (float*)d_out;

  const size_t nX = (size_t)B_ * S_ * D_;  // 8388608
  const size_t nW = (size_t)D_ * D_;       // 4194304
  _Float16* Xh  = (_Float16*)d_ws;
  _Float16* Wqh = Xh + nX;
  _Float16* Wkh = Wqh + nW;
  _Float16* Wvh = Wkh + nW;
  _Float16* Qh  = Wvh + nW;
  _Float16* Kh  = Qh + nX;
  _Float16* Vth = Kh + nX;

  const int thr = 256;
  cvt_f32_f16<<<(int)((nX + thr - 1) / thr), thr, 0, stream>>>(seq, Xh, (int)nX);
  cvt_f32_f16<<<(int)((nW + thr - 1) / thr), thr, 0, stream>>>(Wq, Wqh, (int)nW);
  cvt_f32_f16<<<(int)((nW + thr - 1) / thr), thr, 0, stream>>>(Wk, Wkh, (int)nW);
  cvt_f32_f16<<<(int)((nW + thr - 1) / thr), thr, 0, stream>>>(Wv, Wvh, (int)nW);

  // 64x64 per wave, 128(M) x 256(N) per block
  dim3 gg(D_ / 256, (B_ * S_) / 128);  // (8, 32)
  qkv_gemm_kernel<<<gg, thr, 0, stream>>>(Xh, Wqh, bq, freqs, Qh, 1, 0);
  qkv_gemm_kernel<<<gg, thr, 0, stream>>>(Xh, Wkh, bk, freqs, Kh, 1, 0);
  qkv_gemm_kernel<<<gg, thr, 0, stream>>>(Xh, Wvh, bv, freqs, Vth, 0, 1);

  // 512 blocks: bh (32) x 16 q-blocks; 8 waves/block -> 128 q-tiles of 16
  flash_attn_kernel<<<(B_ * H_) * (S_ / 128), thr, 0, stream>>>(Qh, Kh, Vth,
                                                                mask, out);
}
